// LearntFixedWingDynamics_59545426591883
// MI455X (gfx1250) — matrix operand order repaired and checked by reading, not compiled, not run on hardware
//
#include <hip/hip_runtime.h>
#include <hip/hip_bf16.h>
#include <math.h>

typedef __attribute__((ext_vector_type(16))) _Float16 v16h;
typedef __attribute__((ext_vector_type(8)))  _Float16 h8;
typedef __attribute__((ext_vector_type(8)))  float    v8f;

enum {
  C_g, C_mass, C_rho, C_S, C_c, C_b, C_epsilon,
  C_CL0, C_CL_alpha, C_CL_q, C_CL_del_e,
  C_CD0, C_CD_alpha, C_CD_q, C_CD_del_e,
  C_CY0, C_CY_beta, C_CY_p, C_CY_r, C_CY_del_a, C_CY_del_r,
  C_Cl0, C_Cl_beta, C_Cl_p, C_Cl_r, C_Cl_del_a, C_Cl_del_r,
  C_Cm0, C_Cm_alpha, C_Cm_q, C_Cm_del_e,
  C_Cn0, C_Cn_beta, C_Cn_p, C_Cn_r, C_Cn_del_a, C_Cn_del_r,
  C_COUNT  // 37
};

struct CfgPtrs { const float* p[C_COUNT]; };

// Wave-scope LDS ordering: all LDS traffic here is wave-private. CDNA5 DS ops
// from one wave complete in order at the LDS unit (ISA dep-counter rules), so
// hardware already orders same-wave store->load; we only need to stop the
// compiler from reordering cross-lane-aliasing DS ops, plus a wavefront fence
// so the memory model is respected at IR level. No s_barrier needed.
__device__ __forceinline__ void wave_lds_fence() {
    __builtin_amdgcn_fence(__ATOMIC_ACQ_REL, "wavefront");
    __builtin_amdgcn_wave_barrier();
}

__launch_bounds__(256)
__global__ void fixedwing_wmma_kernel(const float* __restrict__ state,
                                      const float* __restrict__ action,
                                      CfgPtrs cp,
                                      const float* __restrict__ Imat,
                                      const float* __restrict__ W1,
                                      const float* __restrict__ b1,
                                      const float* __restrict__ W2,
                                      const float* __restrict__ b2,
                                      const float* __restrict__ dtp,
                                      float* __restrict__ out,
                                      int B)
{
    // Per-wave private LDS slices (8 waves per block)
    __shared__ __align__(16) _Float16 lds_sa[8][32][16];   // sa tile, f16 (8 KB)
    __shared__ __align__(16) _Float16 lds_h [8][16][64];   // hidden tile, f16 (16 KB)
    __shared__ __align__(16) float    lds_out[8][32][12];  // next_state (+delta) (12 KB)

    const int tid  = threadIdx.x;
    const int w    = tid >> 5;        // wave in block
    const int lane = tid & 31;
    const int half = lane >> 4;       // lane half (K-split in WMMA layouts)
    const int l16  = lane & 15;       // N / M index within 16

    const int gr  = blockIdx.x * 256 + tid;          // this lane's row
    const int grc = (gr < B) ? gr : (B - 1);         // clamped (keep EXEC full)

    // ---------------- load row (coalesced 48B + 16B) ----------------
    const float4* sp = (const float4*)(state + (size_t)grc * 12);
    float4 s0 = sp[0], s1 = sp[1], s2 = sp[2];
    float4 a4 = *(const float4*)(action + (size_t)grc * 4);

    // ---------------- stage sa = [state, action] as f16 ----------------
    {
        h8 lo, hi;
        lo[0]=(_Float16)s0.x; lo[1]=(_Float16)s0.y; lo[2]=(_Float16)s0.z; lo[3]=(_Float16)s0.w;
        lo[4]=(_Float16)s1.x; lo[5]=(_Float16)s1.y; lo[6]=(_Float16)s1.z; lo[7]=(_Float16)s1.w;
        hi[0]=(_Float16)s2.x; hi[1]=(_Float16)s2.y; hi[2]=(_Float16)s2.z; hi[3]=(_Float16)s2.w;
        hi[4]=(_Float16)a4.x; hi[5]=(_Float16)a4.y; hi[6]=(_Float16)a4.z; hi[7]=(_Float16)a4.w;
        *(h8*)&lds_sa[w][lane][0] = lo;
        *(h8*)&lds_sa[w][lane][8] = hi;
    }

    // ---------------- uniform constants ----------------
    float cf[C_COUNT];
    #pragma unroll
    for (int i = 0; i < C_COUNT; ++i) cf[i] = *cp.p[i];
    const float dt = *dtp;

    const float I00=Imat[0], I01=Imat[1], I02=Imat[2];
    const float I10=Imat[3], I11=Imat[4], I12=Imat[5];
    const float I20=Imat[6], I21=Imat[7], I22=Imat[8];
    // inv(I) via adjugate
    const float n00 = I11*I22 - I12*I21, n01 = I02*I21 - I01*I22, n02 = I01*I12 - I02*I11;
    const float n10 = I12*I20 - I10*I22, n11 = I00*I22 - I02*I20, n12 = I02*I10 - I00*I12;
    const float n20 = I10*I21 - I11*I20, n21 = I01*I20 - I00*I21, n22 = I00*I11 - I01*I10;
    const float idet = 1.0f / (I00*n00 + I01*n10 + I02*n20);
    const float v00=n00*idet, v01=n01*idet, v02=n02*idet;
    const float v10=n10*idet, v11=n11*idet, v12=n12*idet;
    const float v20=n20*idet, v21=n21*idet, v22=n22*idet;

    // ---------------- per-lane fp32 physics ----------------
    const float PIf = 3.14159265358979323846f;
    const float AB  = 0.17453292519943295f;  // 10 deg

    const float px=s0.x, py=s0.y, pz=s0.z;
    const float u=s0.w, v=s1.x, w_=s1.y;
    const float phi=s1.z, theta=s1.w, psi=s2.x;
    const float p=s2.y, q=s2.z, r=s2.w;

    const float T   = a4.x * 7.0f;
    const float de  = PIf * (a4.y * 40.0f - 20.0f) / 180.0f;
    const float da  = PIf * (a4.z * 5.0f  - 2.5f ) / 180.0f;
    const float dr  = PIf * (a4.w * 40.0f - 20.0f) / 180.0f;

    const float V     = sqrtf(u*u + v*v + w_*w_);
    const float alpha = fminf(fmaxf(atanf(w_ / u), -AB), AB);
    const float beta  = fminf(fmaxf(atanf(v  / V), -AB), AB);
    const float invV  = 1.0f / V;
    const float c2Vq  = cf[C_c] * 0.5f * invV * q;
    const float b2Vp  = cf[C_b] * 0.5f * invV * p;
    const float b2Vr  = cf[C_b] * 0.5f * invV * r;

    const float CL = cf[C_CL0] + cf[C_CL_alpha]*alpha + cf[C_CL_q]*c2Vq + cf[C_CL_del_e]*de;
    const float CD = cf[C_CD0] + cf[C_CD_alpha]*alpha + cf[C_CD_q]*c2Vq + cf[C_CD_del_e]*de;
    const float CY = cf[C_CY0] + cf[C_CY_beta]*beta + cf[C_CY_p]*b2Vp + cf[C_CY_r]*b2Vr
                   + cf[C_CY_del_a]*da + cf[C_CY_del_r]*dr;
    const float Cl = cf[C_Cl0] + cf[C_Cl_beta]*beta + cf[C_Cl_p]*b2Vp + cf[C_Cl_r]*b2Vr
                   + cf[C_Cl_del_a]*da + cf[C_Cl_del_r]*dr;
    const float Cm = cf[C_Cm0] + cf[C_Cm_alpha]*alpha + cf[C_Cm_q]*c2Vq + cf[C_Cm_del_e]*de;
    const float Cn = cf[C_Cn0] + cf[C_Cn_beta]*beta + cf[C_Cn_p]*b2Vp + cf[C_Cn_r]*b2Vr
                   + cf[C_Cn_del_a]*da + cf[C_Cn_del_r]*dr;

    const float qbarS = 0.5f * cf[C_rho] * V * V * cf[C_S];
    const float Lf = qbarS * CL, Df = qbarS * CD, Yf = qbarS * CY;
    const float lm = qbarS * cf[C_c] * Cl;
    const float mm = qbarS * cf[C_c] * Cm;
    const float nm = qbarS * cf[C_c] * Cn;

    const float sal = sinf(alpha), cal = cosf(alpha);
    const float sbe = sinf(beta),  cbe = cosf(beta);
    // F_aero = R_bw @ [-D, Y, -L]
    const float fax = cal*cbe*(-Df) - cal*sbe*Yf + sal*Lf;
    const float fay = sbe*(-Df) + cbe*Yf;
    const float faz = sal*cbe*(-Df) - sal*sbe*Yf - cal*Lf;

    const float sph=sinf(phi), cph=cosf(phi);
    const float sth=sinf(theta), cth=cosf(theta);
    const float sps=sinf(psi), cps=cosf(psi);

    const float gm  = cf[C_g] * cf[C_mass];
    const float fgx = -sth * gm, fgy = sph*cth*gm, fgz = cph*cth*gm;
    const float eps = cf[C_epsilon];
    const float ftx = T * cosf(eps), ftz = T * sinf(eps);

    const float fx = fax + fgx + ftx;
    const float fy = fay + fgy;
    const float fz = faz + fgz + ftz;

    // pos_dot = M(phi,theta,psi)^T @ vel
    const float m00=cth*cps,               m01=cth*sps,               m02=-sth;
    const float m10=sph*sth*cps - cph*sps, m11=cph*cps + sph*sth*sps, m12=sph*cth;
    const float m20=sph*sps + cph*sth*cps, m21=cph*sth*sps - sph*cps, m22=cph*cth;
    const float pdx = m00*u + m10*v + m20*w_;
    const float pdy = m01*u + m11*v + m21*w_;
    const float pdz = m02*u + m12*v + m22*w_;

    const float invm = 1.0f / cf[C_mass];
    const float udot = fx*invm - (q*w_ - r*v);
    const float vdot = fy*invm - (r*u  - p*w_);
    const float wdot = fz*invm - (p*v  - q*u);

    const float icth = 1.0f / cth;
    const float tth  = sth * icth;
    const float phid = p + sph*tth*q + cph*tth*r;
    const float thed = cph*q - sph*r;
    const float psid = (sph*q + cph*r) * icth;

    const float Iwx = I00*p + I01*q + I02*r;
    const float Iwy = I10*p + I11*q + I12*r;
    const float Iwz = I20*p + I21*q + I22*r;
    const float tx = lm - (q*Iwz - r*Iwy);
    const float ty = mm - (r*Iwx - p*Iwz);
    const float tz = nm - (p*Iwy - q*Iwx);
    const float omdx = v00*tx + v01*ty + v02*tz;
    const float omdy = v10*tx + v11*ty + v12*tz;
    const float omdz = v20*tx + v21*ty + v22*tz;

    // next_state -> LDS (frees registers; deltas accumulated into it later)
    {
        float4* op = (float4*)&lds_out[w][lane][0];
        float4 o0, o1, o2;
        o0.x = px  + dt*pdx;  o0.y = py    + dt*pdy;  o0.z = pz  + dt*pdz;  o0.w = u + dt*udot;
        o1.x = v   + dt*vdot; o1.y = w_    + dt*wdot; o1.z = phi + dt*phid; o1.w = theta + dt*thed;
        o2.x = psi + dt*psid; o2.y = p     + dt*omdx; o2.z = q   + dt*omdy; o2.w = r + dt*omdz;
        op[0] = o0; op[1] = o1; op[2] = o2;
    }

    // ---------------- WMMA B fragments + biases (uniform, L2-hot) ----------------
    // GEMM1: B = W1^T (16x64, K zero-padded to 32). Lane layout (f16 B, 32x16):
    //   lanes 0-15 hold K=0..7 (elems 0-7), lanes 16-31 hold K=8..15 (elems 0-7); elems 8-15 = K16..31 pad->0.
    float bias1[4];
    #pragma unroll
    for (int t = 0; t < 4; ++t) bias1[t] = b1[16*t + l16];
    const float bias2 = (l16 < 12) ? b2[l16] : 0.0f;

    v16h bW1[4];
    #pragma unroll
    for (int t = 0; t < 4; ++t) {
        v16h bf;
        #pragma unroll
        for (int k = 0; k < 16; ++k) bf[k] = (_Float16)0.0f;
        {
            const float* wr = W1 + (size_t)(16*t + l16) * 16 + half*8;   // W1 row n, K half*8..+7
            #pragma unroll
            for (int k = 0; k < 8; ++k) bf[k] = (_Float16)wr[k];
        }
        bW1[t] = bf;
    }
    // GEMM2: B = W2^T (64x12 padded to 64x16), split into two K=32 fragments.
    v16h bW2[2];
    #pragma unroll
    for (int c = 0; c < 2; ++c) {
        v16h bf;
        #pragma unroll
        for (int k = 0; k < 16; ++k) bf[k] = (_Float16)0.0f;
        if (l16 < 12) {
            const float* wr = W2 + (size_t)l16 * 64 + c*32 + half*8;
            #pragma unroll
            for (int k = 0; k < 16; ++k) bf[k] = (_Float16)wr[(k & 8) * 2 + (k & 7)];
        }
        bW2[c] = bf;
    }

    wave_lds_fence();   // sa tile (this wave's slice) visible to this wave's lanes

    // ---------------- two 16-row WMMA tiles per wave ----------------
    for (int t2 = 0; t2 < 2; ++t2) {
        // A fragment of sa (16x32 f16, K=16..31 zero pad):
        //   lanes 0-15: VGPR0-3 = K0..7, VGPR4-7 = K16..23 (pad)
        //   lanes16-31: VGPR0-3 = K8..15, VGPR4-7 = K24..31 (pad)
        union Frag { v16h v; h8 h2[2]; } ua;
        ua.h2[0] = *(const h8*)&lds_sa[w][t2*16 + l16][half * 8];
        h8 z;
        #pragma unroll
        for (int i = 0; i < 8; ++i) z[i] = (_Float16)0.0f;
        ua.h2[1] = z;

        v8f hacc[4];
        #pragma unroll
        for (int t = 0; t < 4; ++t) {
            v8f cc;
            #pragma unroll
            for (int i = 0; i < 8; ++i) cc[i] = bias1[t];   // bias as C operand
            hacc[t] = __builtin_amdgcn_wmma_f32_16x16x32_f16(
                false, ua.v, false, bW1[t], (short)0, cc, false, false);
        }

        // relu -> LDS h tile (row-major 16x64 f16); C layout: M = rr + 8*half, N = 16t+l16
        #pragma unroll
        for (int t = 0; t < 4; ++t) {
            #pragma unroll
            for (int rr = 0; rr < 8; ++rr) {
                lds_h[w][rr + 8*half][16*t + l16] = (_Float16)fmaxf(hacc[t][rr], 0.0f);
            }
        }
        wave_lds_fence();   // h stores -> h fragment loads (same wave, in-order DS)

        // GEMM2: two chained K=32 WMMAs over h (16x64)
        v8f acc2;
        #pragma unroll
        for (int i = 0; i < 8; ++i) acc2[i] = bias2;
        #pragma unroll
        for (int c = 0; c < 2; ++c) {
            union Frag ub;
            ub.h2[0] = *(const h8*)&lds_h[w][l16][c*32 + half*8];        // K base..+7
            ub.h2[1] = *(const h8*)&lds_h[w][l16][c*32 + 16 + half*8];   // K base+16..+23
            acc2 = __builtin_amdgcn_wmma_f32_16x16x32_f16(
                false, ub.v, false, bW2[c], (short)0, acc2, false, false);
        }

        // delta += into next_state tile (unique addresses; ds_add_f32)
        if (l16 < 12) {
            #pragma unroll
            for (int rr = 0; rr < 8; ++rr) {
                atomicAdd(&lds_out[w][t2*16 + rr + 8*half][l16], acc2[rr]);
            }
        }
        wave_lds_fence();   // delta adds -> (next tile's h rewrite) / final row reads
    }

    // ---------------- coalesced output: each lane stores its own 48B row ----------------
    if (gr < B) {
        const float4* dp = (const float4*)&lds_out[w][lane][0];
        float4 o0 = dp[0], o1 = dp[1], o2 = dp[2];
        float4* po = (float4*)(out + (size_t)gr * 12);
        po[0] = o0; po[1] = o1; po[2] = o2;
    }
}

extern "C" void kernel_launch(void* const* d_in, const int* in_sizes, int n_in,
                              void* d_out, int out_size, void* d_ws, size_t ws_size,
                              hipStream_t stream) {
    (void)d_ws; (void)ws_size; (void)out_size;
    const float* state  = (const float*)d_in[0];
    const float* action = (const float*)d_in[1];

    CfgPtrs cp;
    const float *Imat, *W1, *b1, *W2, *b2, *dt;
    if (n_in == 9) {
        // cfg delivered as one contiguous 37-float buffer
        const float* cfg = (const float*)d_in[2];
        for (int i = 0; i < C_COUNT; ++i) cp.p[i] = cfg + i;
        Imat = (const float*)d_in[3];
        W1   = (const float*)d_in[4];
        b1   = (const float*)d_in[5];
        W2   = (const float*)d_in[6];
        b2   = (const float*)d_in[7];
        dt   = (const float*)d_in[8];
    } else {
        // cfg delivered as 37 scalar inputs (n_in == 45)
        for (int i = 0; i < C_COUNT; ++i) cp.p[i] = (const float*)d_in[2 + i];
        const int base = 2 + C_COUNT;
        Imat = (const float*)d_in[base + 0];
        W1   = (const float*)d_in[base + 1];
        b1   = (const float*)d_in[base + 2];
        W2   = (const float*)d_in[base + 3];
        b2   = (const float*)d_in[base + 4];
        dt   = (const float*)d_in[base + 5];
    }

    const int B = in_sizes[0] / 12;
    const int blocks = (B + 255) / 256;
    fixedwing_wmma_kernel<<<blocks, 256, 0, stream>>>(
        state, action, cp, Imat, W1, b1, W2, b2, dt, (float*)d_out, B);
}
